// TCNGenerator_65257733095649
// MI455X (gfx1250) — matrix operand order repaired and checked by reading, not compiled, not run on hardware
//
#include <hip/hip_runtime.h>

typedef __attribute__((ext_vector_type(16))) _Float16 v16h;
typedef __attribute__((ext_vector_type(8)))  _Float16 v8h;
typedef __attribute__((ext_vector_type(8)))  float    v8f;

#define DEV __device__ __forceinline__

static constexpr int BATCH = 32;
static constexpr int TLEN  = 4096;
static constexpr int HID   = 256;
static constexpr int NZ    = 128;
static constexpr int STEPW = 16;
static constexpr int SEQ   = 256;              // TLEN/STEPW
static constexpr int NROW  = BATCH * STEPW;    // 512 folded rows
static constexpr long NPOS = (long)NROW * SEQ; // 131072 attention positions

DEV int lane_id() { return threadIdx.x & 31; }
DEV int wave_id() { return threadIdx.x >> 5; }

DEV v8f wmma_f16(v16h a, v16h b, v8f c) {
  return __builtin_amdgcn_wmma_f32_16x16x32_f16(false, a, false, b, (short)0, c,
                                                false, false);
}

// A fragment: W row-major [M x ld] f16. lane m = m0+(lane&15);
// halves = W[m][k0 + 16*(lane>=16) + 0..15]  (two 16B loads, contiguous)
DEV v16h frag_a(const _Float16* W, int ld, int m0, int k0) {
  int l = lane_id();
  const _Float16* p = W + (long)(m0 + (l & 15)) * ld + k0 + ((l >> 4) << 4);
  v8h lo = *(const v8h*)p;
  v8h hi = *(const v8h*)(p + 8);
  v16h a;
#pragma unroll
  for (int i = 0; i < 8; ++i) { a[i] = lo[i]; a[8 + i] = hi[i]; }
  return a;
}

// B fragment: X row-major [N x ld] f16 (activations [pos][chan]).
// lane n = n0+(lane&15); halves = X[n][k0 + 16*(lane>=16) + 0..15] (contiguous)
DEV v16h frag_b(const _Float16* X, long ld, int n0, int k0) {
  int l = lane_id();
  const _Float16* p = X + (long)(n0 + (l & 15)) * ld + k0 + ((l >> 4) << 4);
  v8h lo = *(const v8h*)p;
  v8h hi = *(const v8h*)(p + 8);
  v16h b;
#pragma unroll
  for (int i = 0; i < 8; ++i) { b[i] = lo[i]; b[8 + i] = hi[i]; }
  return b;
}

// B fragment with per-lane row < 0 -> zeros (causal left padding of convs)
DEV v16h frag_b_guard(const _Float16* X, long ld, int n0, int k0) {
  int l = lane_id();
  int n = n0 + (l & 15);
  v16h b;
  if (n >= 0) {
    const _Float16* p = X + (long)n * ld + k0 + ((l >> 4) << 4);
    v8h lo = *(const v8h*)p;
    v8h hi = *(const v8h*)(p + 8);
#pragma unroll
    for (int i = 0; i < 8; ++i) { b[i] = lo[i]; b[8 + i] = hi[i]; }
  } else {
#pragma unroll
    for (int i = 0; i < 16; ++i) b[i] = (_Float16)0.f;
  }
  return b;
}

// ---------------- weight prep ----------------

__global__ void k_cvt(_Float16* dst, const float* src, int n) {
  for (int i = blockIdx.x * blockDim.x + threadIdx.x; i < n;
       i += gridDim.x * blockDim.x)
    dst[i] = (_Float16)src[i];
}

// (Cout,Cin,2) f32 -> two [Cout x Cin] f16 matrices (tap0 -> t-d, tap1 -> t)
__global__ void k_split2(_Float16* d0, _Float16* d1, const float* src, int n) {
  for (int i = blockIdx.x * blockDim.x + threadIdx.x; i < n;
       i += gridDim.x * blockDim.x) {
    d0[i] = (_Float16)src[2 * i];
    d1[i] = (_Float16)src[2 * i + 1];
  }
}

// z [B,128,T] f32 -> zT [B,T,128] f16 (channel-innermost for B fragments)
__global__ void k_fold_zT(_Float16* dst, const float* src) {
  __shared__ float tile[32][33];
  int b = blockIdx.z;
  int t0 = blockIdx.x * 32, c0 = blockIdx.y * 32;
  const float* s = src + (long)b * NZ * TLEN;
  for (int i = threadIdx.y; i < 32; i += 8)
    tile[i][threadIdx.x] = s[(long)(c0 + i) * TLEN + t0 + threadIdx.x];
  __syncthreads();
  _Float16* d = dst + (long)b * TLEN * NZ;
  for (int i = threadIdx.y; i < 32; i += 8)
    d[(long)(t0 + i) * NZ + c0 + threadIdx.x] = (_Float16)tile[threadIdx.x][i];
}

// ---------------- TCN block 0 (1x1 convs, fused) ----------------
// Y = relu( relu(W2 @ relu(W1 z + b1) + b2) + (Wd z + bd) ), 64 positions/block

__global__ __launch_bounds__(512) void k_tb0(
    const _Float16* __restrict__ zT, const _Float16* __restrict__ W1,
    const _Float16* __restrict__ W2, const _Float16* __restrict__ Wd,
    const float* __restrict__ b1, const float* __restrict__ b2,
    const float* __restrict__ bd, _Float16* __restrict__ Y) {
  constexpr int HLD = HID + 16;
  __shared__ _Float16 Hs[64][HLD];
  long t0 = (long)blockIdx.x * 64;
  int w = wave_id(), l = lane_id();
  const _Float16* zrow = zT + t0 * NZ;
  // phase 1: H = relu(W1 z + b1)   (256 x 64 -> LDS, f16)
  for (int tile = w; tile < 64; tile += 16) {
    int m0 = (tile >> 2) << 4, n0 = (tile & 3) << 4;
    v8f acc = {};
    for (int k0 = 0; k0 < NZ; k0 += 32)
      acc = wmma_f16(frag_a(W1, NZ, m0, k0), frag_b(zrow, NZ, n0, k0), acc);
    int t = n0 + (l & 15), c0 = m0 + ((l >> 4) << 3);
    v8h o;
#pragma unroll
    for (int r = 0; r < 8; ++r) {
      float v = acc[r] + b1[c0 + r];
      o[r] = (_Float16)(v > 0.f ? v : 0.f);
    }
    *(v8h*)&Hs[t][c0] = o;
  }
  __syncthreads();
  // phase 2: W2@H (LDS) and Wd@z (global), combine
  for (int tile = w; tile < 64; tile += 16) {
    int m0 = (tile >> 2) << 4, n0 = (tile & 3) << 4;
    v8f a2 = {}, ad = {};
    for (int k0 = 0; k0 < HID; k0 += 32)
      a2 = wmma_f16(frag_a(W2, HID, m0, k0), frag_b(&Hs[0][0], HLD, n0, k0), a2);
    for (int k0 = 0; k0 < NZ; k0 += 32)
      ad = wmma_f16(frag_a(Wd, NZ, m0, k0), frag_b(zrow, NZ, n0, k0), ad);
    int t = n0 + (l & 15), c0 = m0 + ((l >> 4) << 3);
    v8h o;
#pragma unroll
    for (int r = 0; r < 8; ++r) {
      float h2 = a2[r] + b2[c0 + r];
      h2 = h2 > 0.f ? h2 : 0.f;
      float y = h2 + ad[r] + bd[c0 + r];
      o[r] = (_Float16)(y > 0.f ? y : 0.f);
    }
    *(v8h*)(Y + (t0 + t) * HID + c0) = o;
  }
}

// ---------------- TCN blocks 1..3 (K=2 dilated, fused, identity residual) ----
// H computed for 80 positions (16-pos halo) so conv2 can read H[t-d] locally.

__global__ __launch_bounds__(512) void k_tbk(
    const _Float16* __restrict__ X, const _Float16* __restrict__ W10,
    const _Float16* __restrict__ W11, const _Float16* __restrict__ W20,
    const _Float16* __restrict__ W21, const float* __restrict__ b1,
    const float* __restrict__ b2, _Float16* __restrict__ Y, int dil) {
  constexpr int HLD = HID + 16;
  __shared__ _Float16 Hs[80][HLD];
  int b = blockIdx.x >> 6;           // TLEN/64 = 64 blocks per batch
  int t0 = (blockIdx.x & 63) << 6;
  int w = wave_id(), l = lane_id();
  const _Float16* Xb = X + (long)b * TLEN * HID;
  // phase 1: H = relu(W10 x[t-d] + W11 x[t] + b1) on [t0-16, t0+64)
  for (int tile = w; tile < 80; tile += 16) {
    int m0 = (tile / 5) << 4;
    int pi = tile % 5;
    int tb = t0 - 16 + (pi << 4);
    v8f acc = {};
    for (int k0 = 0; k0 < HID; k0 += 32) {
      acc = wmma_f16(frag_a(W10, HID, m0, k0), frag_b_guard(Xb, HID, tb - dil, k0), acc);
      acc = wmma_f16(frag_a(W11, HID, m0, k0), frag_b_guard(Xb, HID, tb, k0), acc);
    }
    int t = tb + (l & 15);
    int tl = (pi << 4) + (l & 15);
    int c0 = m0 + ((l >> 4) << 3);
    v8h o;
#pragma unroll
    for (int r = 0; r < 8; ++r) {
      float v = acc[r] + b1[c0 + r];
      v = v > 0.f ? v : 0.f;
      o[r] = (t < 0) ? (_Float16)0.f : (_Float16)v;  // conv padding is zero
    }
    *(v8h*)&Hs[tl][c0] = o;
  }
  __syncthreads();
  // phase 2: Y = relu( relu(W20 H[t-d] + W21 H[t] + b2) + x[t] )
  for (int tile = w; tile < 64; tile += 16) {
    int m0 = (tile >> 2) << 4, n0 = (tile & 3) << 4;
    int hb = 16 + n0;
    v8f acc = {};
    for (int k0 = 0; k0 < HID; k0 += 32) {
      acc = wmma_f16(frag_a(W20, HID, m0, k0), frag_b(&Hs[0][0], HLD, hb - dil, k0), acc);
      acc = wmma_f16(frag_a(W21, HID, m0, k0), frag_b(&Hs[0][0], HLD, hb, k0), acc);
    }
    int t = t0 + n0 + (l & 15), c0 = m0 + ((l >> 4) << 3);
    v8h res = *(const v8h*)(Xb + (long)t * HID + c0);
    v8h o;
#pragma unroll
    for (int r = 0; r < 8; ++r) {
      float v = acc[r] + b2[c0 + r];
      v = v > 0.f ? v : 0.f;
      float y = v + (float)res[r];
      o[r] = (_Float16)(y > 0.f ? y : 0.f);
    }
    *(v8h*)(Y + ((long)b * TLEN + t) * HID + c0) = o;
  }
}

// ---------------- QKV projection over folded view ----------------
// x[row][s][c] = Z[b][s*16+step][c]; stride between s rows is 16*HID halves.

__global__ __launch_bounds__(256) void k_qkv(
    const _Float16* __restrict__ Z, const _Float16* __restrict__ W,
    const float* __restrict__ bias, _Float16* __restrict__ q,
    _Float16* __restrict__ k, _Float16* __restrict__ v) {
  int row = blockIdx.x >> 4;
  int s0 = (blockIdx.x & 15) << 4;
  int b = row >> 4, st = row & 15;
  int w = wave_id(), l = lane_id();
  const _Float16* base = Z + ((long)b * TLEN + st) * HID;
  for (int tile = w; tile < 48; tile += 8) {   // 768 out channels / 16
    int o0 = tile << 4;
    v8f acc = {};
    for (int k0 = 0; k0 < HID; k0 += 32)
      acc = wmma_f16(frag_a(W, HID, o0, k0),
                     frag_b(base, (long)STEPW * HID, s0, k0), acc);
    int s = s0 + (l & 15);
    int oc = o0 + ((l >> 4) << 3);
    _Float16* dst;
    int c;
    if (oc < HID)          { dst = q; c = oc; }
    else if (oc < 2 * HID) { dst = k; c = oc - HID; }
    else                   { dst = v; c = oc - 2 * HID; }
    v8h o;
#pragma unroll
    for (int r = 0; r < 8; ++r) o[r] = (_Float16)(acc[r] + bias[oc + r]);
    *(v8h*)(dst + ((long)row * SEQ + s) * HID + c) = o;
  }
}

// ---------------- causal window-8 attention core (tiny FLOPs, VALU) ---------

__global__ __launch_bounds__(256) void k_attn(
    const _Float16* __restrict__ q, const _Float16* __restrict__ k,
    const _Float16* __restrict__ v, _Float16* __restrict__ o) {
  long row = blockIdx.x;
  const float scale = 0.125f;  // 1/sqrt(64)
  for (int pair = threadIdx.x; pair < SEQ * 4; pair += 256) {
    int s = pair >> 2, h = pair & 3;
    const _Float16* qp = q + (row * SEQ + s) * HID + h * 64;
    int jmin = s - 7 > 0 ? s - 7 : 0;
    int nj = s - jmin + 1;
    float sc[8];
    float m = -3.0e38f;
    for (int j = 0; j < nj; ++j) {
      const _Float16* kp = k + (row * SEQ + jmin + j) * HID + h * 64;
      float d = 0.f;
      for (int c = 0; c < 64; ++c) d += (float)qp[c] * (float)kp[c];
      d *= scale;
      sc[j] = d;
      m = d > m ? d : m;
    }
    float sum = 0.f;
    for (int j = 0; j < nj; ++j) { sc[j] = __expf(sc[j] - m); sum += sc[j]; }
    float inv = 1.f / sum;
    for (int j = 0; j < nj; ++j) sc[j] *= inv;
    const _Float16* vp = v + (row * SEQ + jmin) * HID + h * 64;
    _Float16* op = o + (row * SEQ + s) * HID + h * 64;
    for (int c = 0; c < 64; ++c) {
      float a = 0.f;
      for (int j = 0; j < nj; ++j) a += sc[j] * (float)vp[(long)j * HID + c];
      op[c] = (_Float16)a;
    }
  }
}

// ---------------- out-proj + residual + LayerNorm1 ----------------

__global__ __launch_bounds__(256) void k_oproj_ln(
    const _Float16* __restrict__ O, const _Float16* __restrict__ W,
    const float* __restrict__ bias, const _Float16* __restrict__ Z,
    const float* __restrict__ g, const float* __restrict__ bb,
    _Float16* __restrict__ X1) {
  __shared__ float xs[32][HID + 1];
  long n0 = (long)blockIdx.x * 32;
  int row = (int)(n0 >> 8), s0 = (int)(n0 & 255);
  int b = row >> 4, st = row & 15;
  int w = wave_id(), l = lane_id();
  for (int tile = w; tile < 32; tile += 8) {
    int m0 = (tile >> 1) << 4, p0 = (tile & 1) << 4;
    v8f acc = {};
    for (int k0 = 0; k0 < HID; k0 += 32)
      acc = wmma_f16(frag_a(W, HID, m0, k0), frag_b(O + n0 * HID, HID, p0, k0), acc);
    int p = p0 + (l & 15), c0 = m0 + ((l >> 4) << 3);
    int t = (s0 + p) * STEPW + st;
    v8h xr = *(const v8h*)(Z + ((long)b * TLEN + t) * HID + c0);
#pragma unroll
    for (int r = 0; r < 8; ++r)
      xs[p][c0 + r] = acc[r] + bias[c0 + r] + (float)xr[r];
  }
  __syncthreads();
  for (int p = w * 4; p < w * 4 + 4; ++p) {
    float s = 0.f, ss = 0.f;
#pragma unroll
    for (int i = 0; i < 8; ++i) { float x = xs[p][l * 8 + i]; s += x; ss += x * x; }
    for (int off = 16; off > 0; off >>= 1) {
      s += __shfl_xor(s, off, 32);
      ss += __shfl_xor(ss, off, 32);
    }
    float mean = s * (1.f / HID);
    float var = ss * (1.f / HID) - mean * mean;
    float inv = rsqrtf(var + 1e-5f);
    v8h o;
#pragma unroll
    for (int i = 0; i < 8; ++i) {
      int c = l * 8 + i;
      o[i] = (_Float16)((xs[p][c] - mean) * inv * g[c] + bb[c]);
    }
    *(v8h*)(X1 + (n0 + p) * HID + l * 8) = o;
  }
}

// ---------------- FFN up-projection + exact GELU ----------------

__global__ __launch_bounds__(512) void k_ffn1(
    const _Float16* __restrict__ X1, const _Float16* __restrict__ W,
    const float* __restrict__ bias, _Float16* __restrict__ H) {
  long n0 = (long)blockIdx.x * 64;
  int w = wave_id(), l = lane_id();
  for (int tile = w; tile < 128; tile += 16) {  // 512 out ch x 64 pos
    int m0 = (tile >> 2) << 4, p0 = (tile & 3) << 4;
    v8f acc = {};
    for (int k0 = 0; k0 < HID; k0 += 32)
      acc = wmma_f16(frag_a(W, HID, m0, k0), frag_b(X1 + n0 * HID, HID, p0, k0), acc);
    int p = p0 + (l & 15), c0 = m0 + ((l >> 4) << 3);
    v8h o;
#pragma unroll
    for (int r = 0; r < 8; ++r) {
      float x = acc[r] + bias[c0 + r];
      o[r] = (_Float16)(0.5f * x * (1.f + erff(x * 0.70710678118f)));
    }
    *(v8h*)(H + (n0 + p) * 512 + c0) = o;
  }
}

// ---------------- FFN down-proj + residual + LN2 + final 256->1 dot ---------

__global__ __launch_bounds__(256) void k_ffn2_ln_out(
    const _Float16* __restrict__ H, const _Float16* __restrict__ W,
    const float* __restrict__ bias, const _Float16* __restrict__ X1,
    const float* __restrict__ g, const float* __restrict__ bb,
    const float* __restrict__ lw, const float* __restrict__ lb,
    float* __restrict__ out) {
  __shared__ float xs[32][HID + 1];
  long n0 = (long)blockIdx.x * 32;
  int row = (int)(n0 >> 8), s0 = (int)(n0 & 255);
  int b = row >> 4, st = row & 15;
  int w = wave_id(), l = lane_id();
  for (int tile = w; tile < 32; tile += 8) {
    int m0 = (tile >> 1) << 4, p0 = (tile & 1) << 4;
    v8f acc = {};
    for (int k0 = 0; k0 < 512; k0 += 32)
      acc = wmma_f16(frag_a(W, 512, m0, k0), frag_b(H + n0 * 512, 512, p0, k0), acc);
    int p = p0 + (l & 15), c0 = m0 + ((l >> 4) << 3);
    v8h xr = *(const v8h*)(X1 + (n0 + p) * HID + c0);
#pragma unroll
    for (int r = 0; r < 8; ++r)
      xs[p][c0 + r] = acc[r] + bias[c0 + r] + (float)xr[r];
  }
  __syncthreads();
  for (int p = w * 4; p < w * 4 + 4; ++p) {
    float s = 0.f, ss = 0.f;
#pragma unroll
    for (int i = 0; i < 8; ++i) { float x = xs[p][l * 8 + i]; s += x; ss += x * x; }
    for (int off = 16; off > 0; off >>= 1) {
      s += __shfl_xor(s, off, 32);
      ss += __shfl_xor(ss, off, 32);
    }
    float mean = s * (1.f / HID), var = ss * (1.f / HID) - mean * mean;
    float inv = rsqrtf(var + 1e-5f);
    float pd = 0.f;
#pragma unroll
    for (int i = 0; i < 8; ++i) {
      int c = l * 8 + i;
      float nv = (xs[p][c] - mean) * inv * g[c] + bb[c];
      pd += nv * lw[c];
    }
    for (int off = 16; off > 0; off >>= 1) pd += __shfl_xor(pd, off, 32);
    if (l == 0) {
      int t = (s0 + p) * STEPW + st;  // unfold: out[b][0][t]
      out[(long)b * TLEN + t] = pd + lb[0];
    }
  }
}

// ---------------- host orchestration ----------------

extern "C" void kernel_launch(void* const* d_in, const int* in_sizes, int n_in,
                              void* d_out, int out_size, void* d_ws,
                              size_t ws_size, hipStream_t stream) {
  (void)in_sizes; (void)n_in; (void)out_size; (void)ws_size;
  const float* z        = (const float*)d_in[0];
  const float* tb0_b1   = (const float*)d_in[2];
  const float* tb0_b2   = (const float*)d_in[4];
  const float* tb0_bd   = (const float*)d_in[6];
  const float* tb_w1[4] = {nullptr, (const float*)d_in[7],  (const float*)d_in[11], (const float*)d_in[15]};
  const float* tb_b1[4] = {nullptr, (const float*)d_in[8],  (const float*)d_in[12], (const float*)d_in[16]};
  const float* tb_w2[4] = {nullptr, (const float*)d_in[9],  (const float*)d_in[13], (const float*)d_in[17]};
  const float* tb_b2[4] = {nullptr, (const float*)d_in[10], (const float*)d_in[14], (const float*)d_in[18]};
  const float* attn_in_b  = (const float*)d_in[20];
  const float* attn_out_b = (const float*)d_in[22];
  const float* ln1_g = (const float*)d_in[23];
  const float* ln1_b = (const float*)d_in[24];
  const float* ln2_g = (const float*)d_in[25];
  const float* ln2_b = (const float*)d_in[26];
  const float* ffn_b1 = (const float*)d_in[28];
  const float* ffn_b2 = (const float*)d_in[30];
  const float* last_w = (const float*)d_in[31];
  const float* last_b = (const float*)d_in[32];
  float* out = (float*)d_out;

  char* ws = (char*)d_ws;
  const size_t MB = 1ull << 20;
  _Float16* actA = (_Float16*)(ws);             // 64 MB ping
  _Float16* actB = (_Float16*)(ws + 64 * MB);   // 64 MB pong (zT lives here first)
  _Float16* qbuf = (_Float16*)(ws + 128 * MB);  // 32 MB
  _Float16* kbuf = (_Float16*)(ws + 160 * MB);  // 32 MB
  _Float16* vbuf = (_Float16*)(ws + 192 * MB);  // 32 MB
  _Float16* hbuf = (_Float16*)(ws + 128 * MB);  // 64 MB, reuses q/k/v after attn
  _Float16* x1   = (_Float16*)(ws + 224 * MB);  // 32 MB
  _Float16* wptr = (_Float16*)(ws + 256 * MB);  // ~3 MB of f16 weights
  auto walloc = [&](size_t n) { _Float16* p = wptr; wptr += n; return p; };

  _Float16* w0_1 = walloc(256 * 128);
  _Float16* w0_2 = walloc(256 * 256);
  _Float16* w0_d = walloc(256 * 128);
  _Float16 *w1a[4], *w1b[4], *w2a[4], *w2b[4];
  for (int i = 1; i <= 3; ++i) {
    w1a[i] = walloc(256 * 256); w1b[i] = walloc(256 * 256);
    w2a[i] = walloc(256 * 256); w2b[i] = walloc(256 * 256);
  }
  _Float16* wqkv = walloc(768 * 256);
  _Float16* wout = walloc(256 * 256);
  _Float16* wf1  = walloc(512 * 256);
  _Float16* wf2  = walloc(256 * 512);

  auto cvt = [&](_Float16* d, const float* s, int n) {
    int gsz = (n + 255) / 256; if (gsz > 2048) gsz = 2048;
    k_cvt<<<gsz, 256, 0, stream>>>(d, s, n);
  };
  cvt(w0_1, (const float*)d_in[1], 256 * 128);
  cvt(w0_2, (const float*)d_in[3], 256 * 256);
  cvt(w0_d, (const float*)d_in[5], 256 * 128);
  for (int i = 1; i <= 3; ++i) {
    k_split2<<<256, 256, 0, stream>>>(w1a[i], w1b[i], tb_w1[i], 256 * 256);
    k_split2<<<256, 256, 0, stream>>>(w2a[i], w2b[i], tb_w2[i], 256 * 256);
  }
  cvt(wqkv, (const float*)d_in[19], 768 * 256);
  cvt(wout, (const float*)d_in[21], 256 * 256);
  cvt(wf1,  (const float*)d_in[27], 512 * 256);
  cvt(wf2,  (const float*)d_in[29], 256 * 512);

  // z [B,128,T] f32 -> zT [B,T,128] f16
  k_fold_zT<<<dim3(TLEN / 32, NZ / 32, BATCH), dim3(32, 8), 0, stream>>>(actB, z);

  // TCN stack (ping-pong)
  const int nblk = BATCH * TLEN / 64;  // 2048
  k_tb0<<<nblk, 512, 0, stream>>>(actB, w0_1, w0_2, w0_d, tb0_b1, tb0_b2, tb0_bd, actA);
  k_tbk<<<nblk, 512, 0, stream>>>(actA, w1a[1], w1b[1], w2a[1], w2b[1], tb_b1[1], tb_b2[1], actB, 1);
  k_tbk<<<nblk, 512, 0, stream>>>(actB, w1a[2], w1b[2], w2a[2], w2b[2], tb_b1[2], tb_b2[2], actA, 2);
  k_tbk<<<nblk, 512, 0, stream>>>(actA, w1a[3], w1b[3], w2a[3], w2b[3], tb_b1[3], tb_b2[3], actB, 4);

  // Attention block over folded view (Z = actB)
  k_qkv<<<NROW * 16, 256, 0, stream>>>(actB, wqkv, attn_in_b, qbuf, kbuf, vbuf);
  k_attn<<<NROW, 256, 0, stream>>>(qbuf, kbuf, vbuf, actA);
  k_oproj_ln<<<(int)(NPOS / 32), 256, 0, stream>>>(actA, wout, attn_out_b, actB,
                                                   ln1_g, ln1_b, x1);
  k_ffn1<<<(int)(NPOS / 64), 512, 0, stream>>>(x1, wf1, ffn_b1, hbuf);
  k_ffn2_ln_out<<<(int)(NPOS / 32), 256, 0, stream>>>(hbuf, wf2, ffn_b2, x1,
                                                      ln2_g, ln2_b, last_w,
                                                      last_b, out);
}
